// AttentionNetwork_13073880449169
// MI455X (gfx1250) — matrix operand order repaired and checked by reading, compiled, not run on hardware
//
#include <hip/hip_runtime.h>

typedef __attribute__((ext_vector_type(16))) _Float16 v16h;
typedef __attribute__((ext_vector_type(8)))  float    v8f;

#define T_SEQ 64
#define D_MOD 128
#define XS 136   // f16 LDS stride for x (pad: avoid 64-bank conflicts)
#define KVS 132  // f32 LDS stride for k/v buffer

// ---------------------------------------------------------------------------
// Pre-kernel: Wk, Wv (f32 row-major [d,e]) -> f16 transposed [e,d] in workspace
// so each WMMA B-fragment lane reads 16 consecutive-K f16 (32B contiguous).
// ---------------------------------------------------------------------------
__global__ __launch_bounds__(256) void convert_wt_kernel(
    const float* __restrict__ Wk, const float* __restrict__ Wv,
    _Float16* __restrict__ WkT, _Float16* __restrict__ WvT) {
  int i = blockIdx.x * 256 + threadIdx.x;          // i = d*128 + e, 16384 total
  int d = i >> 7, e = i & 127;
  WkT[e * D_MOD + d] = (_Float16)Wk[i];
  WvT[e * D_MOD + d] = (_Float16)Wv[i];
}

// ---------------------------------------------------------------------------
// 64x128 = xf16[64x128] @ WT^T   (WT is [e,d] f16), f32 accumulate via WMMA.
// 8 waves: wave = N-tile (8 x 16 cols), loop 4 M-tiles x 4 K-steps.
// ---------------------------------------------------------------------------
__device__ inline void gemm_x_wt(const _Float16* __restrict__ WT,
                                 const _Float16* __restrict__ xf16,
                                 float* __restrict__ kv,
                                 int wave, int lane) {
  const int lo = lane & 15;
  const int hi = lane >> 4;
  const int nt = wave;                              // N tile 0..7

  // B fragments: N = nt*16+lo, K = ks*32 + hi*16 + (0..15) contiguous
  v16h bfr[4];
  const _Float16* wrow = WT + (nt * 16 + lo) * D_MOD + hi * 16;
#pragma unroll
  for (int ks = 0; ks < 4; ++ks) {
    union { unsigned u[8]; v16h h; } bu;
    const unsigned* p = (const unsigned*)(wrow + ks * 32);
#pragma unroll
    for (int j = 0; j < 8; ++j) bu.u[j] = p[j];
    bfr[ks] = bu.h;
  }

#pragma unroll
  for (int mt = 0; mt < 4; ++mt) {
    v8f acc = {};
#pragma unroll
    for (int ks = 0; ks < 4; ++ks) {
      // A fragment: row = mt*16+lo, K = ks*32 + hi*16 + (0..15) contiguous
      union { unsigned u[8]; v16h h; } au;
      const _Float16* xr = xf16 + (mt * 16 + lo) * XS + ks * 32 + hi * 16;
      const unsigned* p0 = (const unsigned*)xr;
      const unsigned* p1 = (const unsigned*)(xr + 8);
#pragma unroll
      for (int j = 0; j < 4; ++j) { au.u[j] = p0[j]; au.u[4 + j] = p1[j]; }
      acc = __builtin_amdgcn_wmma_f32_16x16x32_f16(
          false, au.h, false, bfr[ks], (short)0, acc, false, false);
    }
    // C layout: element r -> row = mt*16 + r + hi*8, col = nt*16 + lo
#pragma unroll
    for (int r = 0; r < 8; ++r)
      kv[(mt * 16 + r + hi * 8) * KVS + nt * 16 + lo] = acc[r];
  }
}

// ---------------------------------------------------------------------------
// Main kernel: one workgroup (8 wave32) per batch element.
// Mask structure: only query row 63 is live; rows 0..62 -> uniform attention
// = column mean of V; FF GEMM collapses to 2 distinct rows.
// ---------------------------------------------------------------------------
__global__ __launch_bounds__(256) void attn_fused_kernel(
    const float* __restrict__ inputs, const float* __restrict__ pos_emb,
    const float* __restrict__ Wq, const float* __restrict__ W_lin,
    const _Float16* __restrict__ WkT, const _Float16* __restrict__ WvT,
    float* __restrict__ out) {
  __shared__ _Float16 xf16[T_SEQ * XS];   // 17408 B
  __shared__ float    kv[T_SEQ * KVS];    // 33792 B (K, then reused for V)
  __shared__ float    x63[D_MOD];
  __shared__ float    qlast[D_MOD];
  __shared__ float    sc[8 * T_SEQ];
  __shared__ float    attnw[8 * T_SEQ];
  __shared__ float    vmean[D_MOD];
  __shared__ float    out63[D_MOD];
  __shared__ float    ffu[D_MOD];
  __shared__ float    ff63[D_MOD];

  const int tid  = threadIdx.x;
  const int lane = tid & 31;
  const int wave = tid >> 5;
  const int b    = blockIdx.x;
  const float* inb = inputs + (size_t)b * (T_SEQ * D_MOD);

  // Phase A: x = inputs + pos_emb  ->  f16 LDS (+ keep f32 row 63)
  for (int i = tid; i < T_SEQ * D_MOD; i += 256) {
    int t = i >> 7, e = i & 127;
    float xv = inb[i] + pos_emb[i];
    xf16[t * XS + e] = (_Float16)xv;
    if (t == T_SEQ - 1) x63[e] = xv;
  }
  __syncthreads();

  // Phase B: K = x @ Wk  (WMMA, f32 accum into kv)
  gemm_x_wt(WkT, xf16, kv, wave, lane);
  __syncthreads();

  // q row 63 (exact f32): qlast[e] = sum_d x63[d] * Wq[d,e]
  if (tid < D_MOD) {
    float s = 0.f;
    for (int d = 0; d < D_MOD; ++d) s += x63[d] * Wq[d * D_MOD + tid];
    qlast[tid] = s;
  }
  __syncthreads();

  // Scores for the live row: s[h,t] = (q63_h . k[t]_h) / sqrt(T)
  for (int si = tid; si < 8 * T_SEQ; si += 256) {
    int h = si >> 6, t = si & 63;
    float s = 0.f;
#pragma unroll
    for (int i = 0; i < 16; ++i)
      s += qlast[h * 16 + i] * kv[t * KVS + h * 16 + i];
    sc[si] = s * 0.125f;                          // 1/sqrt(64)
  }
  __syncthreads();

  // Softmax per head (tiny: 8 heads x 64)
  if (tid < 8) {
    float m = -3.4e38f;
    for (int t = 0; t < T_SEQ; ++t) m = fmaxf(m, sc[tid * T_SEQ + t]);
    float ssum = 0.f;
    for (int t = 0; t < T_SEQ; ++t) {
      float e = __expf(sc[tid * T_SEQ + t] - m);
      attnw[tid * T_SEQ + t] = e;
      ssum += e;
    }
    float inv = 1.f / ssum;
    for (int t = 0; t < T_SEQ; ++t) attnw[tid * T_SEQ + t] *= inv;
  }
  __syncthreads();

  // Phase C: V = x @ Wv  (WMMA) — overwrites kv (K no longer needed)
  gemm_x_wt(WvT, xf16, kv, wave, lane);
  __syncthreads();

  // vmean = column mean of V (== out rows 0..62); out63 = attn @ V (row 63)
  if (tid < D_MOD) {
    int e = tid, h = e >> 4;
    float s = 0.f, a = 0.f;
    for (int t = 0; t < T_SEQ; ++t) {
      float vv = kv[t * KVS + e];
      s += vv;
      a += attnw[h * T_SEQ + t] * vv;
    }
    vmean[e] = s * 0.015625f;                     // 1/64 exact
    out63[e] = a;
  }
  __syncthreads();

  // FF for the 2 distinct rows: ff[e] = relu(sum_d row[d]*W_lin[e,d]) + row[e]
  {
    int e = tid & 127;
    const float* src = (tid < 128) ? vmean : out63;
    const float* wl = W_lin + e * D_MOD;
    float s = 0.f;
    for (int d = 0; d < D_MOD; ++d) s += src[d] * wl[d];
    float r = fmaxf(s, 0.f) + src[e];
    if (tid < 128) ffu[e] = r; else ff63[e] = r;
  }
  __syncthreads();

  // Phase E: result = ff_row + x (recompute x from inputs+pos_emb; L2 hit)
  float* ob = out + (size_t)b * (T_SEQ * D_MOD);
  for (int i = tid; i < T_SEQ * D_MOD; i += 256) {
    int t = i >> 7, e = i & 127;
    float xv = inb[i] + pos_emb[i];
    float f = (t == T_SEQ - 1) ? ff63[e] : ffu[e];
    ob[i] = f + xv;
  }
}

// ---------------------------------------------------------------------------
extern "C" void kernel_launch(void* const* d_in, const int* in_sizes, int n_in,
                              void* d_out, int out_size, void* d_ws, size_t ws_size,
                              hipStream_t stream) {
  const float* inputs  = (const float*)d_in[0];
  const float* pos_emb = (const float*)d_in[1];
  const float* Wq      = (const float*)d_in[2];
  const float* Wk      = (const float*)d_in[3];
  const float* Wv      = (const float*)d_in[4];
  const float* W_lin   = (const float*)d_in[5];
  float* out = (float*)d_out;

  _Float16* WkT = (_Float16*)d_ws;                 // 128*128 f16 = 32 KB
  _Float16* WvT = WkT + D_MOD * D_MOD;             // + 32 KB

  const int B = in_sizes[0] / (T_SEQ * D_MOD);     // 4096

  convert_wt_kernel<<<dim3((D_MOD * D_MOD) / 256), dim3(256), 0, stream>>>(
      Wk, Wv, WkT, WvT);
  attn_fused_kernel<<<dim3(B), dim3(256), 0, stream>>>(
      inputs, pos_emb, Wq, W_lin, WkT, WvT, out);
}
